// GPT_model_27650999451990
// MI455X (gfx1250) — matrix operand order repaired and checked by reading, not compiled, not run on hardware
//
#include <hip/hip_runtime.h>

// ---------------- model constants ----------------
#define BATCH   64
#define T_SEQ   128
#define N_EMBD  384
#define N_LAYER 16
#define N_HEAD  16
#define HEAD_D  24
#define HEAD_DP 32          // padded head dim so K%32==0
#define D_FF    1536
#define VOCAB   32000
#define M_TOK   (BATCH * T_SEQ)      // 8192 rows of activations
#define BH      (BATCH * N_HEAD)     // 1024 attention batches
#define KQV_N   (3 * N_EMBD)         // 1152

// Use GLOBAL_LOAD_ASYNC_TO_LDS for the B-panel staging (ASYNCcnt path).
#define ASYNC_STAGE 1

typedef __attribute__((ext_vector_type(16))) __bf16 v16bf;
typedef __attribute__((ext_vector_type(8)))  __bf16 v8bf;
typedef __attribute__((ext_vector_type(8)))  float  v8f;

#define OUT_F32     0
#define OUT_F32_ACC 1
#define OUT_BF16    2

static __device__ __forceinline__ unsigned short f32_to_bf16(float f) {
    unsigned int u = __float_as_uint(f);
    unsigned int r = u + 0x7FFFu + ((u >> 16) & 1u);   // round-to-nearest-even
    return (unsigned short)(r >> 16);
}

// =====================================================================
// Batched bf16 WMMA GEMM:  C[M,N] (+)= A[M,K] * Bt[N,K]^T + bias
//   A  row-major [M,K] bf16
//   Bt row-major [N,K] bf16 (B pre-transposed -> contiguous frag loads)
// Block = 256 threads = 8 waves.
//   Block tile: (MT*128) rows x (NT*16) cols; wave tile MT*16 x NT*16.
// Caller guarantees: M % (MT*128)==0, N % (NT*16)==0, K % 32 == 0.
// B panel staged into LDS with GLOBAL_LOAD_ASYNC_TO_LDS_B128 (no VGPR
// round-trip, ASYNCcnt-tracked), double-buffered; main loop manually
// unrolled 2x (ping-pong A-register sets + LDS buffers -> no swap movs).
// =====================================================================
template<int NT, int MT>
__global__ __launch_bounds__(256)
void gemm_bf16_wmma(const unsigned short* __restrict__ Au,
                    const unsigned short* __restrict__ Btu,
                    void* __restrict__ Cout,
                    const float* __restrict__ bias,
                    int M, int N, int K, int ldc,
                    long long strideA, long long strideB, long long strideC,
                    int outMode, int doGelu)
{
    constexpr int BN = NT * 16;                 // block cols
    __shared__ __align__(16) unsigned short Bs[2][BN][40];  // 32 elems + 8 pad

    const int lane = threadIdx.x & 31;
    const int wave = threadIdx.x >> 5;
    const int half = lane >> 4;      // 0: lanes 0-15, 1: lanes 16-31
    const int l16  = lane & 15;

    const long long zb = blockIdx.z;
    const __bf16* A = (const __bf16*)Au + zb * strideA;
    const unsigned short* Bt = Btu + zb * strideB;

    const int n0 = blockIdx.x * BN;
    const int m0 = blockIdx.y * (MT * 128) + wave * (MT * 16);

    v8f acc[MT][NT];
#pragma unroll
    for (int mt = 0; mt < MT; mt++)
#pragma unroll
        for (int nt = 0; nt < NT; nt++)
#pragma unroll
            for (int i = 0; i < 8; i++) acc[mt][nt][i] = 0.0f;

    // A fragment pointers (per-lane: row m, two 16B chunks at kb, kb+16)
    const __bf16* aptr[MT];
#pragma unroll
    for (int mt = 0; mt < MT; mt++)
        aptr[mt] = A + (long long)(m0 + mt * 16 + l16) * K + half * 8;

    // B staging: thread -> (row, 16B chunk) of the [BN x 32] panel.
    // All 256 lanes stage (wrap for NT==2 -> benign duplicate writes),
    // so there is no divergent branch around the staging op.
    const int chunk = threadIdx.x & (BN * 4 - 1);
    const int crow  = chunk >> 2;
    const int coff  = (chunk & 3) * 8;          // element offset in row
    const unsigned short* bsrc = Bt + (long long)(n0 + crow) * K + coff;

    const int nsteps = K >> 5;

    // step helpers ----------------------------------------------------
    auto loadA = [&](int s, v16bf (&af)[MT]) {
        const int k = s << 5;
#pragma unroll
        for (int mt = 0; mt < MT; mt++) {
            *(v8bf*)&af[mt]       = *(const v8bf*)(aptr[mt] + k);
            *((v8bf*)&af[mt] + 1) = *(const v8bf*)(aptr[mt] + k + 16);
        }
    };
    auto stageB = [&](int s, int buf) {
        const int k = s << 5;
#if ASYNC_STAGE
        // memory -> LDS without VGPR round-trip; tracked by ASYNCcnt.
        // Generic shared pointers: low 32 bits == LDS byte offset.
        unsigned lds = (unsigned)(size_t)&Bs[buf][crow][coff];
        unsigned long long g = (unsigned long long)(size_t)(bsrc + k);
        asm volatile("global_load_async_to_lds_b128 %0, %1, off"
                     :: "v"(lds), "v"(g) : "memory");
#else
        *(uint4*)&Bs[buf][crow][coff] = *(const uint4*)(bsrc + k);
#endif
        if (s + 2 < nsteps)
            __builtin_prefetch((const void*)(bsrc + k + 64), 0, 3);
    };
    auto asyncFence = []() {
#if ASYNC_STAGE
        asm volatile("s_wait_asynccnt 0x0" ::: "memory");
#endif
    };
    auto compute = [&](int buf, const v16bf (&af)[MT]) {
        // load ALL B frags first -> batched ds loads, then all WMMAs
        v8bf b0[NT], b1[NT];
#pragma unroll
        for (int nt = 0; nt < NT; nt++) {
            const unsigned short* brow = &Bs[buf][nt * 16 + l16][half * 8];
            b0[nt] = *(const v8bf*)brow;
            b1[nt] = *(const v8bf*)(brow + 16);
        }
#pragma unroll
        for (int nt = 0; nt < NT; nt++) {
            v16bf bf_;
            *(v8bf*)&bf_       = b0[nt];
            *((v8bf*)&bf_ + 1) = b1[nt];
#pragma unroll
            for (int mt = 0; mt < MT; mt++)
                acc[mt][nt] = __builtin_amdgcn_wmma_f32_16x16x32_bf16(
                    false, af[mt], false, bf_, (short)0, acc[mt][nt], false, false);
        }
    };
    // ------------------------------------------------------------------

    v16bf afE[MT], afO[MT];

    // prologue: stage step 0 into buf0, A frags for step 0 into set E
    stageB(0, 0);
    loadA(0, afE);
    asyncFence();
    __syncthreads();

    int s = 0;
    for (; s + 1 < nsteps; s += 2) {
        // even step s: consume buf0 / set E; prefetch s+1 -> buf1 / set O
        stageB(s + 1, 1);
        loadA(s + 1, afO);
        compute(0, afE);
        asyncFence();
        __syncthreads();

        // odd step s+1: consume buf1 / set O; prefetch s+2 -> buf0 / set E
        if (s + 2 < nsteps) {
            stageB(s + 2, 0);
            loadA(s + 2, afE);
        }
        compute(1, afO);
        asyncFence();
        __syncthreads();
    }
    if (s < nsteps) {            // odd nsteps tail (buf0 / set E already ready)
        compute(0, afE);
    }

    // ---- epilogue: C layout col = lane&15, row = vgpr + 8*(lane>>4) ----
#pragma unroll
    for (int nt = 0; nt < NT; nt++) {
        const int gc = n0 + nt * 16 + l16;
        const float bv = bias ? bias[gc] : 0.0f;
#pragma unroll
        for (int mt = 0; mt < MT; mt++) {
            const int gr0 = m0 + mt * 16 + half * 8;
#pragma unroll
            for (int i = 0; i < 8; i++) {
                float v = acc[mt][nt][i] + bv;
                if (doGelu) v = 0.5f * v * (1.0f + erff(v * 0.70710678118654752f));
                long long idx = zb * strideC + (long long)(gr0 + i) * ldc + gc;
                if (outMode == OUT_BF16)          ((unsigned short*)Cout)[idx] = f32_to_bf16(v);
                else if (outMode == OUT_F32_ACC)  ((float*)Cout)[idx] += v;
                else                              ((float*)Cout)[idx]  = v;
            }
        }
    }
}

// =====================================================================
// Embedding: x[bt, c] = tok_emb[tokens[bt], c] + pos_emb[t, c]
// =====================================================================
__global__ void embed_kernel(const int* __restrict__ tokens,
                             const float* __restrict__ tok_emb,
                             const float* __restrict__ pos_emb,
                             float* __restrict__ x)
{
    int idx = blockIdx.x * blockDim.x + threadIdx.x;
    if (idx >= M_TOK * N_EMBD) return;
    int bt = idx / N_EMBD, c = idx % N_EMBD, t = bt % T_SEQ;
    x[idx] = tok_emb[(long long)tokens[bt] * N_EMBD + c] + pos_emb[t * N_EMBD + c];
}

// =====================================================================
// LayerNorm over rows of 384; 128 threads/row, 3 elems/thread.
// =====================================================================
__global__ __launch_bounds__(128)
void ln_kernel(const float* __restrict__ xin,
               float* __restrict__ xout_f32,
               unsigned short* __restrict__ out_b16,
               const float* __restrict__ g, const float* __restrict__ b)
{
    const int r = blockIdx.x, tid = threadIdx.x;
    const float* row = xin + (long long)r * N_EMBD;
    float v0 = row[tid], v1 = row[tid + 128], v2 = row[tid + 256];

    __shared__ float red[4];
    float s = v0 + v1 + v2;
#pragma unroll
    for (int off = 16; off >= 1; off >>= 1) s += __shfl_xor(s, off, 32);
    if ((tid & 31) == 0) red[tid >> 5] = s;
    __syncthreads();
    s = red[0] + red[1] + red[2] + red[3];
    float mu = s * (1.0f / N_EMBD);

    float d0 = v0 - mu, d1 = v1 - mu, d2 = v2 - mu;
    float q = d0 * d0 + d1 * d1 + d2 * d2;
#pragma unroll
    for (int off = 16; off >= 1; off >>= 1) q += __shfl_xor(q, off, 32);
    __syncthreads();
    if ((tid & 31) == 0) red[tid >> 5] = q;
    __syncthreads();
    q = red[0] + red[1] + red[2] + red[3];
    float rstd = rsqrtf(q * (1.0f / N_EMBD) + 1e-5f);

#pragma unroll
    for (int j = 0; j < 3; j++) {
        int c = tid + j * 128;
        float vv = (j == 0 ? v0 : (j == 1 ? v1 : v2));
        float o = (vv - mu) * rstd * g[c] + b[c];
        long long idx = (long long)r * N_EMBD + c;
        if (xout_f32) xout_f32[idx] = o;
        if (out_b16)  out_b16[idx] = f32_to_bf16(o);
    }
}

// =====================================================================
// Causal softmax over rows of scores w[(b,h,t), s]; emits bf16 A matrix.
// =====================================================================
__global__ __launch_bounds__(128)
void softmax_kernel(const float* __restrict__ w, unsigned short* __restrict__ a_b16)
{
    const long long row = blockIdx.x;           // (b*H + h)*T + t
    const int t = (int)(row % T_SEQ);
    const int s = threadIdx.x;
    const long long base = row * T_SEQ;

    float v = (s <= t) ? w[base + s] : -3.0e38f;
    __shared__ float red[4];
    float m = v;
#pragma unroll
    for (int off = 16; off >= 1; off >>= 1) m = fmaxf(m, __shfl_xor(m, off, 32));
    if ((threadIdx.x & 31) == 0) red[threadIdx.x >> 5] = m;
    __syncthreads();
    m = fmaxf(fmaxf(red[0], red[1]), fmaxf(red[2], red[3]));

    float e = (s <= t) ? __expf(v - m) : 0.0f;
    float sum = e;
#pragma unroll
    for (int off = 16; off >= 1; off >>= 1) sum += __shfl_xor(sum, off, 32);
    __syncthreads();
    if ((threadIdx.x & 31) == 0) red[threadIdx.x >> 5] = sum;
    __syncthreads();
    sum = red[0] + red[1] + red[2] + red[3];

    a_b16[base + s] = f32_to_bf16(e / sum);
}

// =====================================================================
// Split kqv_cat [M,1152] fp32 into padded bf16 tensors:
//   kpad,qpad: [b,h,t,32] (d>=24 zero),  vt: [b,h,32,t] (transposed)
// =====================================================================
__global__ void split_heads_kernel(const float* __restrict__ kqv,
                                   unsigned short* __restrict__ kpad,
                                   unsigned short* __restrict__ qpad,
                                   unsigned short* __restrict__ vt)
{
    int idx = blockIdx.x * blockDim.x + threadIdx.x;   // M*H*32
    if (idx >= M_TOK * N_HEAD * HEAD_DP) return;
    int d  = idx & (HEAD_DP - 1);
    int h  = (idx >> 5) & (N_HEAD - 1);
    int bt = idx >> 9;
    int b = bt >> 7, t = bt & (T_SEQ - 1);
    int bh = b * N_HEAD + h;

    float kv = 0.f, qv = 0.f, vv = 0.f;
    if (d < HEAD_D) {
        int col = h * HEAD_D + d;
        const float* rowp = kqv + (long long)bt * KQV_N;
        kv = rowp[col];
        qv = rowp[N_EMBD + col];
        vv = rowp[2 * N_EMBD + col];
    }
    long long pidx = ((long long)bh * T_SEQ + t) * HEAD_DP + d;
    kpad[pidx] = f32_to_bf16(kv);
    qpad[pidx] = f32_to_bf16(qv);
    vt[((long long)bh * HEAD_DP + d) * T_SEQ + t] = f32_to_bf16(vv);
}

// =====================================================================
// Merge o_pad [b,h,t,32] fp32 -> o_b16 [bt, 384] bf16 (concat heads)
// =====================================================================
__global__ void merge_heads_kernel(const float* __restrict__ o_pad,
                                   unsigned short* __restrict__ o_b16)
{
    int idx = blockIdx.x * blockDim.x + threadIdx.x;   // M*384
    if (idx >= M_TOK * N_EMBD) return;
    int col = idx % N_EMBD, bt = idx / N_EMBD;
    int h = col / HEAD_D, d = col % HEAD_D;
    int b = bt >> 7, t = bt & (T_SEQ - 1);
    int bh = b * N_HEAD + h;
    o_b16[idx] = f32_to_bf16(o_pad[((long long)bh * T_SEQ + t) * HEAD_DP + d]);
}

// =====================================================================
// Weight converters (fp32 [K,N] -> bf16 transposed [N,K])
// =====================================================================
__global__ void conv_transpose_kernel(const float* __restrict__ in,
                                      unsigned short* __restrict__ out,
                                      int N, int K)
{
    long long idx = (long long)blockIdx.x * blockDim.x + threadIdx.x;
    if (idx >= (long long)N * K) return;
    int n = (int)(idx / K), k = (int)(idx % K);
    out[idx] = f32_to_bf16(in[(long long)k * N + n]);
}

// Wk/Wq/Wv [H, C, D] per layer -> wkqv_t [1152, 384] bf16
__global__ void conv_qkv_kernel(const float* __restrict__ Wk,
                                const float* __restrict__ Wq,
                                const float* __restrict__ Wv,
                                unsigned short* __restrict__ out)
{
    int idx = blockIdx.x * blockDim.x + threadIdx.x;   // 1152*384
    if (idx >= KQV_N * N_EMBD) return;
    int n = idx / N_EMBD, c = idx % N_EMBD;
    int sec = n / N_EMBD, nn = n % N_EMBD;
    int h = nn / HEAD_D, d = nn % HEAD_D;
    const float* W = (sec == 0) ? Wk : (sec == 1 ? Wq : Wv);
    out[idx] = f32_to_bf16(W[((long long)h * N_EMBD + c) * HEAD_D + d]);
}

// =====================================================================
// Host-side orchestration
// =====================================================================
extern "C" void kernel_launch(void* const* d_in, const int* in_sizes, int n_in,
                              void* d_out, int out_size, void* d_ws, size_t ws_size,
                              hipStream_t stream)
{
    const int*   tokens  = (const int*)  d_in[0];
    const float* tok_emb = (const float*)d_in[1];
    const float* pos_emb = (const float*)d_in[2];
    const float* Wk      = (const float*)d_in[3];
    const float* Wq      = (const float*)d_in[4];
    const float* Wv      = (const float*)d_in[5];
    const float* Wproj   = (const float*)d_in[6];
    const float* bproj   = (const float*)d_in[7];
    const float* W1      = (const float*)d_in[8];
    const float* b1      = (const float*)d_in[9];
    const float* W2      = (const float*)d_in[10];
    const float* b2      = (const float*)d_in[11];
    const float* ln1_g   = (const float*)d_in[12];
    const float* ln1_b   = (const float*)d_in[13];
    const float* ln2_g   = (const float*)d_in[14];
    const float* ln2_b   = (const float*)d_in[15];
    const float* lnf_g   = (const float*)d_in[16];
    const float* lnf_b   = (const float*)d_in[17];
    const float* Wout    = (const float*)d_in[18];
    const float* bout    = (const float*)d_in[19];
    float* logits = (float*)d_out;

    // ---- carve scratch ----
    char* ws = (char*)d_ws;
    size_t off = 0;
    auto alloc = [&](size_t bytes) -> void* {
        void* p = ws + off;
        off += (bytes + 255) & ~(size_t)255;
        return p;
    };
    float*          x       = (float*)         alloc((size_t)M_TOK * N_EMBD * 4);
    unsigned short* y_b16   = (unsigned short*)alloc((size_t)M_TOK * N_EMBD * 2);
    unsigned short* x_b16   = (unsigned short*)alloc((size_t)M_TOK * N_EMBD * 2);
    float*          kqv_cat = (float*)         alloc((size_t)M_TOK * KQV_N * 4);
    unsigned short* k_pad   = (unsigned short*)alloc((size_t)BH * T_SEQ * HEAD_DP * 2);
    unsigned short* q_pad   = (unsigned short*)alloc((size_t)BH * T_SEQ * HEAD_DP * 2);
    unsigned short* v_t     = (unsigned short*)alloc((size_t)BH * HEAD_DP * T_SEQ * 2);
    float*          scores  = (float*)         alloc((size_t)BH * T_SEQ * T_SEQ * 4);
    unsigned short* a_b16   = (unsigned short*)alloc((size_t)BH * T_SEQ * T_SEQ * 2);
    float*          o_pad   = (float*)         alloc((size_t)BH * T_SEQ * HEAD_DP * 4);
    unsigned short* o_b16   = (unsigned short*)alloc((size_t)M_TOK * N_EMBD * 2);
    unsigned short* h_b16   = (unsigned short*)alloc((size_t)M_TOK * D_FF * 2);
    unsigned short* wkqv_t  = (unsigned short*)alloc((size_t)KQV_N * N_EMBD * 2);
    unsigned short* wproj_t = (unsigned short*)alloc((size_t)N_EMBD * N_EMBD * 2);
    unsigned short* w1_t    = (unsigned short*)alloc((size_t)D_FF * N_EMBD * 2);
    unsigned short* w2_t    = (unsigned short*)alloc((size_t)N_EMBD * D_FF * 2);
    unsigned short* wout_t  = (unsigned short*)alloc((size_t)VOCAB * N_EMBD * 2);

    // NT/MT dispatch: big GEMMs <4,2>, scores <4,1>, AV <2,1>
    auto gemm = [&](const unsigned short* A, const unsigned short* Bt, void* C,
                    const float* bias, int M, int N, int K, int ldc,
                    long long sA, long long sB, long long sC, int nbatch,
                    int outMode, int gelu, int NT, int MT) {
        dim3 grid(N / (NT * 16), M / (MT * 128), nbatch);
        if (NT == 4 && MT == 2)
            gemm_bf16_wmma<4, 2><<<grid, 256, 0, stream>>>(A, Bt, C, bias, M, N, K, ldc,
                                                           sA, sB, sC, outMode, gelu);
        else if (NT == 4 && MT == 1)
            gemm_bf16_wmma<4, 1><<<grid, 256, 0, stream>>>(A, Bt, C, bias, M, N, K, ldc,
                                                           sA, sB, sC, outMode, gelu);
        else
            gemm_bf16_wmma<2, 1><<<grid, 256, 0, stream>>>(A, Bt, C, bias, M, N, K, ldc,
                                                           sA, sB, sC, outMode, gelu);
    };

    // ---- embedding ----
    {
        int n = M_TOK * N_EMBD;
        embed_kernel<<<(n + 255) / 256, 256, 0, stream>>>(tokens, tok_emb, pos_emb, x);
    }

    for (int l = 0; l < N_LAYER; l++) {
        const float* Wk_l = Wk + (size_t)l * N_HEAD * N_EMBD * HEAD_D;
        const float* Wq_l = Wq + (size_t)l * N_HEAD * N_EMBD * HEAD_D;
        const float* Wv_l = Wv + (size_t)l * N_HEAD * N_EMBD * HEAD_D;

        // weight conversion (transpose to [N,K] bf16)
        {
            int n = KQV_N * N_EMBD;
            conv_qkv_kernel<<<(n + 255) / 256, 256, 0, stream>>>(Wk_l, Wq_l, Wv_l, wkqv_t);
            n = N_EMBD * N_EMBD;
            conv_transpose_kernel<<<(n + 255) / 256, 256, 0, stream>>>(
                Wproj + (size_t)l * N_EMBD * N_EMBD, wproj_t, N_EMBD, N_EMBD);
            n = D_FF * N_EMBD;
            conv_transpose_kernel<<<(n + 255) / 256, 256, 0, stream>>>(
                W1 + (size_t)l * N_EMBD * D_FF, w1_t, D_FF, N_EMBD);
            n = N_EMBD * D_FF;
            conv_transpose_kernel<<<(n + 255) / 256, 256, 0, stream>>>(
                W2 + (size_t)l * D_FF * N_EMBD, w2_t, N_EMBD, D_FF);
        }

        // ln1: y = LN(x) -> bf16 only (x preserved as residual)
        ln_kernel<<<M_TOK, 128, 0, stream>>>(x, (float*)nullptr, y_b16,
                                             ln1_g + l * N_EMBD, ln1_b + l * N_EMBD);

        // kqv = y @ Wkqv  -> fp32 [M, 1152]
        gemm(y_b16, wkqv_t, kqv_cat, nullptr, M_TOK, KQV_N, N_EMBD, KQV_N,
             0, 0, 0, 1, OUT_F32, 0, 4, 2);

        // split into padded bf16 heads (k,q row-major; v transposed)
        {
            int n = M_TOK * N_HEAD * HEAD_DP;
            split_heads_kernel<<<(n + 255) / 256, 256, 0, stream>>>(kqv_cat, k_pad, q_pad, v_t);
        }

        // scores[t,s] = k[t,:] . q[s,:]  (reference computes K @ Q^T, no scale)
        gemm(k_pad, q_pad, scores, nullptr, T_SEQ, T_SEQ, HEAD_DP, T_SEQ,
             (long long)T_SEQ * HEAD_DP, (long long)T_SEQ * HEAD_DP,
             (long long)T_SEQ * T_SEQ, BH, OUT_F32, 0, 4, 1);

        // causal softmax -> bf16 attention matrix
        softmax_kernel<<<BH * T_SEQ, 128, 0, stream>>>(scores, a_b16);

        // o = a @ v   (Bt = v^T [32,128])
        gemm(a_b16, v_t, o_pad, nullptr, T_SEQ, HEAD_DP, T_SEQ, HEAD_DP,
             (long long)T_SEQ * T_SEQ, (long long)HEAD_DP * T_SEQ,
             (long long)T_SEQ * HEAD_DP, BH, OUT_F32, 0, 2, 1);

        // concat heads -> bf16 [M, 384]
        {
            int n = M_TOK * N_EMBD;
            merge_heads_kernel<<<(n + 255) / 256, 256, 0, stream>>>(o_pad, o_b16);
        }

        // x += o @ Wproj + bproj   (fused residual via accumulate epilogue)
        gemm(o_b16, wproj_t, x, bproj + l * N_EMBD, M_TOK, N_EMBD, N_EMBD, N_EMBD,
             0, 0, 0, 1, OUT_F32_ACC, 0, 4, 2);

        // ln2: x = LN(x) in place, also bf16 copy for FFN input
        ln_kernel<<<M_TOK, 128, 0, stream>>>(x, x, x_b16,
                                             ln2_g + l * N_EMBD, ln2_b + l * N_EMBD);

        // h = gelu(x @ W1 + b1) -> bf16 directly
        gemm(x_b16, w1_t, h_b16, b1 + l * D_FF, M_TOK, D_FF, N_EMBD, D_FF,
             0, 0, 0, 1, OUT_BF16, 1, 4, 2);

        // x += h @ W2 + b2
        gemm(h_b16, w2_t, x, b2 + l * N_EMBD, M_TOK, N_EMBD, D_FF, N_EMBD,
             0, 0, 0, 1, OUT_F32_ACC, 0, 4, 2);
    }

    // final LN -> bf16
    ln_kernel<<<M_TOK, 128, 0, stream>>>(x, (float*)nullptr, x_b16, lnf_g, lnf_b);

    // logits = x @ Wout + bout  (fp32 to d_out)
    {
        long long n = (long long)VOCAB * N_EMBD;
        conv_transpose_kernel<<<(int)((n + 255) / 256), 256, 0, stream>>>(
            Wout, wout_t, VOCAB, N_EMBD);
    }
    gemm(x_b16, wout_t, logits, bout, M_TOK, VOCAB, N_EMBD, VOCAB,
         0, 0, 0, 1, OUT_F32, 0, 4, 2);
}
